// SelfAttention_2662879724011
// MI455X (gfx1250) — matrix operand order repaired and checked by reading, not compiled, not run on hardware
//
#include <hip/hip_runtime.h>
#include <hip/hip_bf16.h>
#include <math.h>

// ---------------- problem constants ----------------
#define BATCH     2
#define SEQ       2048
#define DIMC      1024
#define NHEAD     16
#define HDIM      64
#define NROWS     (BATCH * SEQ)        // 4096
#define KSTEPS    (DIMC / 32)          // 32

typedef __attribute__((ext_vector_type(16))) _Float16 v16h;
typedef __attribute__((ext_vector_type(8)))  _Float16 h8;
typedef __attribute__((ext_vector_type(4)))  _Float16 h4;
typedef __attribute__((ext_vector_type(8)))  float    v8f;
typedef __attribute__((ext_vector_type(4)))  unsigned u32x4;
typedef __attribute__((ext_vector_type(8)))  int      i32x8;
typedef __attribute__((ext_vector_type(4)))  int      i32x4;

union Frag16 { v16h v; h8 h[2]; };

#if __has_builtin(__builtin_amdgcn_tensor_load_to_lds) && \
    __has_builtin(__builtin_amdgcn_s_wait_tensorcnt)
#define HAVE_TDM 1
#else
#define HAVE_TDM 0
#endif

__device__ __forceinline__ unsigned lds_off_u32(const void* p) {
    // flat shared address: [63:32]=aperture, [31:0]=LDS byte offset
    return (unsigned)(size_t)p;
}
__device__ __forceinline__ void compiler_mem_barrier() {
    asm volatile("" ::: "memory");
}

#if HAVE_TDM
// ---------------- Tensor Data Mover: generic 2D tile of f16 ----------------
// Tensor: [tensor_d1, stride0(>=tensor_d0)] f16 row-major; tile (tile_d1 rows x
// tile_d0 elems) starting at gtile -> LDS row-major-contiguous at lds_byte_off.
__device__ __forceinline__ void tdm_load_tile(unsigned lds_byte_off,
                                              const _Float16* gtile,
                                              unsigned tensor_d0, unsigned tensor_d1,
                                              unsigned tile_d0,   unsigned tile_d1,
                                              unsigned stride0) {
    unsigned long long ga = (unsigned long long)(size_t)gtile;
    u32x4 g0;
    g0[0] = 1u;                                    // count=1, user mode, no gather
    g0[1] = lds_byte_off;                          // lds_addr
    g0[2] = (unsigned)ga;                          // global_addr[31:0]
    g0[3] = (unsigned)(ga >> 32) | 0x80000000u;    // global_addr[56:32] | type=2
    i32x8 g1;
    g1[0] = 0x00010000;                            // data_size=1 (2 bytes)
    g1[1] = (int)(tensor_d0 << 16);                // tensor_dim0[15:0]
    g1[2] = (int)((tensor_d0 >> 16) | (tensor_d1 << 16));
    g1[3] = (int)((tensor_d1 >> 16) | (tile_d0 << 16));
    g1[4] = (int)tile_d1;                          // tile_dim1 (tile_dim2 = 0)
    g1[5] = (int)stride0;                          // tensor_dim0_stride
    g1[6] = 0;
    g1[7] = 0;
    i32x4 z4 = (i32x4)0;
#if __clang_major__ >= 23
    i32x8 z8 = (i32x8)0;
    __builtin_amdgcn_tensor_load_to_lds(g0, g1, z4, z4, z8, 0);
#else
    __builtin_amdgcn_tensor_load_to_lds(g0, g1, z4, z4, 0);
#endif
}
#else
// ---------------- fallback: per-lane async global->LDS ----------------
__device__ __forceinline__ void async_b128(unsigned lds_off, const _Float16* gptr) {
    asm volatile("global_load_async_to_lds_b128 %0, %1, off"
                 :: "v"(lds_off), "v"(gptr) : "memory");
}
__device__ __forceinline__ void wait_async3() {
    asm volatile("s_wait_asynccnt 0x3" ::: "memory");
}
__device__ __forceinline__ void wait_async0() {
    asm volatile("s_wait_asynccnt 0x0" ::: "memory");
}
#endif

// ---------------- f32 -> f16 convert (x4 vectorized) ----------------
__global__ void ngpt_f32_to_f16(const float4* __restrict__ in,
                                h4* __restrict__ out, int n4) {
    int i = blockIdx.x * blockDim.x + threadIdx.x;
    if (i < n4) {
        float4 f = in[i];
        h4 o = { (_Float16)f.x, (_Float16)f.y, (_Float16)f.z, (_Float16)f.w };
        out[i] = o;
    }
}

// ---------------- shared 128x64x(K=1024) GEMM mainloop ----------------
// y = A @ W.T ; block = 256 threads = 8 waves; wave w owns rows [row0g+16w, +16),
// cols [col0g, +64). Double-buffered LDS tiles staged by the Tensor Data Mover
// (wave 0 issues, TENSORcnt waits, barrier publishes), WMMA from LDS fragments.
__device__ __forceinline__ void gemm_tile_128x64(const _Float16* __restrict__ ga,
                                                 unsigned ga_rows,
                                                 const _Float16* __restrict__ gb,
                                                 int row0g, int col0g,
                                                 v8f acc[4]) {
    __shared__ __align__(16) _Float16 Ash[2][128 * 32];
    __shared__ __align__(16) _Float16 Bsh[2][64 * 32];
    const int t    = threadIdx.x;
    const int lane = t & 31;
    const int wave = t >> 5;
    const int n    = lane & 15;
    const int hi   = lane >> 4;

#if HAVE_TDM
    const bool producer = (wave == 0);
    auto issue = [&](int step) {
        const int k0  = step * 32;
        const int buf = step & 1;
        tdm_load_tile(lds_off_u32(&Ash[buf][0]), ga + (size_t)row0g * DIMC + k0,
                      DIMC, ga_rows, 32u, 128u, DIMC);
        tdm_load_tile(lds_off_u32(&Bsh[buf][0]), gb + (size_t)col0g * DIMC + k0,
                      DIMC, DIMC, 32u, 64u, DIMC);
    };
    if (producer) issue(0);
#else
    (void)ga_rows;
    auto issue = [&](int step) {
        const int k0  = step * 32;
        const int buf = step & 1;
        const unsigned aBase = lds_off_u32(&Ash[buf][0]);
        const unsigned bBase = lds_off_u32(&Bsh[buf][0]);
        { int c = t;       int r = c >> 2, kk = (c & 3) * 8;
          async_b128(aBase + (unsigned)(r * 32 + kk) * 2,
                     ga + (size_t)(row0g + r) * DIMC + k0 + kk); }
        { int c = t + 256; int r = c >> 2, kk = (c & 3) * 8;
          async_b128(aBase + (unsigned)(r * 32 + kk) * 2,
                     ga + (size_t)(row0g + r) * DIMC + k0 + kk); }
        { int c = t;       int r = c >> 2, kk = (c & 3) * 8;
          async_b128(bBase + (unsigned)(r * 32 + kk) * 2,
                     gb + (size_t)(col0g + r) * DIMC + k0 + kk); }
    };
    issue(0);
#endif

    for (int s = 0; s < KSTEPS; ++s) {
#if HAVE_TDM
        if (producer) {
            if (s + 1 < KSTEPS) { issue(s + 1); __builtin_amdgcn_s_wait_tensorcnt(2); }
            else                {               __builtin_amdgcn_s_wait_tensorcnt(0); }
            compiler_mem_barrier();
        }
#else
        if (s + 1 < KSTEPS) { issue(s + 1); wait_async3(); }
        else                {               wait_async0(); }
#endif
        __syncthreads();
        const int buf = s & 1;
        Frag16 a;
        const _Float16* arow = &Ash[buf][(wave * 16 + n) * 32 + 16 * hi];
        a.h[0] = *(const h8*)arow;
        a.h[1] = *(const h8*)(arow + 8);
        Frag16 b[4];
#pragma unroll
        for (int f = 0; f < 4; ++f) {
            const _Float16* brow = &Bsh[buf][(16 * f + n) * 32 + 16 * hi];
            b[f].h[0] = *(const h8*)brow;
            b[f].h[1] = *(const h8*)(brow + 8);
        }
#pragma unroll
        for (int f = 0; f < 4; ++f)
            acc[f] = __builtin_amdgcn_wmma_f32_16x16x32_f16(
                false, a.v, false, b[f].v, (short)0, acc[f], false, false);
        __syncthreads();
    }
}

// ---------------- Q/K GEMM + RoPE + cosine-norm epilogue ----------------
__global__ __launch_bounds__(256)
void ngpt_gemm_qk_rope(const _Float16* __restrict__ x16,
                       const _Float16* __restrict__ w16,
                       const float* __restrict__ sinb,
                       const float* __restrict__ cosb,
                       const float* __restrict__ s_qk,
                       _Float16* __restrict__ out) {
    const int lane = threadIdx.x & 31;
    const int wave = threadIdx.x >> 5;
    const int n    = lane & 15;
    const int hi   = lane >> 4;
    const int row0 = blockIdx.x * 128 + wave * 16;
    const int h    = blockIdx.y;
    const int col0 = h * HDIM;

    v8f acc[4] = {};
    gemm_tile_128x64(x16, NROWS, w16, blockIdx.x * 128, col0, acc);

#pragma unroll
    for (int j = 0; j < 8; ++j) {
        int r = row0 + 8 * hi + j;
        int s = r & (SEQ - 1);
        int b = r >> 11;
        float vals[4];
#pragma unroll
        for (int f = 0; f < 4; ++f) vals[f] = acc[f][j];
        // RoPE: pair (d, d+32) == fragments (f, f+2)
#pragma unroll
        for (int f = 0; f < 2; ++f) {
            int d = 16 * f + n;
            float c1 = cosb[s * HDIM + d],      s1 = sinb[s * HDIM + d];
            float c2 = cosb[s * HDIM + d + 32], s2 = sinb[s * HDIM + d + 32];
            float x1 = vals[f], x2 = vals[f + 2];
            vals[f]     = x1 * c1 - x2 * s1;
            vals[f + 2] = x2 * c2 + x1 * s2;
        }
        float ss = vals[0]*vals[0] + vals[1]*vals[1] + vals[2]*vals[2] + vals[3]*vals[3];
        ss += __shfl_xor(ss, 1);
        ss += __shfl_xor(ss, 2);
        ss += __shfl_xor(ss, 4);
        ss += __shfl_xor(ss, 8);
        float inv = 1.0f / fmaxf(sqrtf(ss), 1e-6f);
        _Float16* orow = out + ((size_t)(b * NHEAD + h) * SEQ + s) * HDIM;
#pragma unroll
        for (int f = 0; f < 4; ++f) {
            int d = 16 * f + n;
            float eff = s_qk[col0 + d] * 32.0f;   // s_qk * sqrt(DIM)
            orow[d] = (_Float16)(vals[f] * inv * eff);
        }
    }
}

// ---------------- V GEMM, stores transposed [b, h, d, s] ----------------
__global__ __launch_bounds__(256)
void ngpt_gemm_v(const _Float16* __restrict__ x16,
                 const _Float16* __restrict__ w16,
                 _Float16* __restrict__ outT) {
    const int lane = threadIdx.x & 31;
    const int wave = threadIdx.x >> 5;
    const int n    = lane & 15;
    const int hi   = lane >> 4;
    const int row0 = blockIdx.x * 128 + wave * 16;
    const int h    = blockIdx.y;

    v8f acc[4] = {};
    gemm_tile_128x64(x16, NROWS, w16, blockIdx.x * 128, h * HDIM, acc);

#pragma unroll
    for (int j = 0; j < 8; ++j) {
        int r = row0 + 8 * hi + j;
        int s = r & (SEQ - 1);
        int b = r >> 11;
        _Float16* base = outT + (size_t)(b * NHEAD + h) * HDIM * SEQ;
#pragma unroll
        for (int f = 0; f < 4; ++f) {
            int d = 16 * f + n;
            base[(size_t)d * SEQ + s] = (_Float16)acc[f][j];
        }
    }
}

// ---------------- output projection GEMM (f32 out) ----------------
__global__ __launch_bounds__(256)
void ngpt_gemm_out(const _Float16* __restrict__ a16,
                   const _Float16* __restrict__ w16,
                   float* __restrict__ out) {
    const int lane = threadIdx.x & 31;
    const int wave = threadIdx.x >> 5;
    const int n    = lane & 15;
    const int hi   = lane >> 4;
    const int row0 = blockIdx.x * 128 + wave * 16;
    const int col0 = blockIdx.y * 64;

    v8f acc[4] = {};
    gemm_tile_128x64(a16, NROWS, w16, blockIdx.x * 128, col0, acc);

#pragma unroll
    for (int j = 0; j < 8; ++j) {
        int r = row0 + 8 * hi + j;
#pragma unroll
        for (int f = 0; f < 4; ++f)
            out[(size_t)r * DIMC + col0 + 16 * f + n] = acc[f][j];
    }
}

// ---------------- Flash attention (1 wave / 16-query tile) ----------------
// K/V tiles staged per 32-key chunk by the TDM, double-buffered; single-wave
// block so TENSORcnt wait alone publishes the LDS data to the consumers.
__global__ __launch_bounds__(32)
void ngpt_flash_attn(const _Float16* __restrict__ q16,
                     const _Float16* __restrict__ k16,
                     const _Float16* __restrict__ vT,
                     _Float16* __restrict__ attn_out) {
#if HAVE_TDM
    __shared__ __align__(16) _Float16 Ksh[2][32 * HDIM];   // [key][d]
    __shared__ __align__(16) _Float16 Vsh[2][HDIM * 32];   // [d][key]
#endif
    __shared__ __align__(16) _Float16 Plds[16 * 32];
    const int lane = threadIdx.x & 31;
    const int n    = lane & 15;
    const int hi   = lane >> 4;
    const int qt   = blockIdx.x;
    const int bh   = blockIdx.y;
    const int b    = bh >> 4;
    const int h    = bh & 15;
    const int q0   = qt * 16;

    const _Float16* Q  = q16 + (size_t)bh * SEQ * HDIM;
    const _Float16* K  = k16 + (size_t)bh * SEQ * HDIM;
    const _Float16* Vt = vT  + (size_t)bh * HDIM * SEQ;

    Frag16 aq[2];
#pragma unroll
    for (int t = 0; t < 2; ++t) {
        const _Float16* ap = Q + (size_t)(q0 + n) * HDIM + 32 * t + 16 * hi;
        aq[t].h[0] = *(const h8*)ap;
        aq[t].h[1] = *(const h8*)(ap + 8);
    }

    v8f o[4] = {};
    float mrow[8], lrow[8];
#pragma unroll
    for (int j = 0; j < 8; ++j) { mrow[j] = -3.0e38f; lrow[j] = 0.0f; }

    const int nch = (q0 + 16 + 31) >> 5;
#if HAVE_TDM
    auto issue = [&](int ch) {
        const int k0  = ch * 32;
        const int buf = ch & 1;
        // K tile: 32 key-rows x 64 halves from [SEQ, HDIM]
        tdm_load_tile(lds_off_u32(&Ksh[buf][0]), K + (size_t)k0 * HDIM,
                      HDIM, SEQ, HDIM, 32u, HDIM);
        // V tile: 64 d-rows x 32 key-cols from [HDIM, SEQ]
        tdm_load_tile(lds_off_u32(&Vsh[buf][0]), Vt + k0,
                      SEQ, HDIM, 32u, HDIM, SEQ);
    };
    issue(0);
#endif
    for (int ch = 0; ch < nch; ++ch) {
        const int k0 = ch * 32;
#if HAVE_TDM
        if (ch + 1 < nch) { issue(ch + 1); __builtin_amdgcn_s_wait_tensorcnt(2); }
        else              {               __builtin_amdgcn_s_wait_tensorcnt(0); }
        compiler_mem_barrier();
        const int buf = ch & 1;
#endif
        v8f sfr[2] = {};
#pragma unroll
        for (int g = 0; g < 2; ++g) {
#pragma unroll
            for (int t = 0; t < 2; ++t) {
                Frag16 bk;
#if HAVE_TDM
                const _Float16* bp = &Ksh[buf][(16 * g + n) * HDIM + 32 * t + 16 * hi];
#else
                const _Float16* bp = K + (size_t)(k0 + 16 * g + n) * HDIM + 32 * t + 16 * hi;
#endif
                bk.h[0] = *(const h8*)bp;
                bk.h[1] = *(const h8*)(bp + 8);
                sfr[g] = __builtin_amdgcn_wmma_f32_16x16x32_f16(
                    false, aq[t].v, false, bk.v, (short)0, sfr[g], false, false);
            }
        }
#pragma unroll
        for (int j = 0; j < 8; ++j) {
            int qrow = q0 + 8 * hi + j;
            float v0 = sfr[0][j] * 8.0f;         // * sqrt(HDIM)
            float v1 = sfr[1][j] * 8.0f;
            if (k0 + n      > qrow) v0 = -3.0e38f;
            if (k0 + 16 + n > qrow) v1 = -3.0e38f;
            float mx = fmaxf(v0, v1);
            mx = fmaxf(mx, __shfl_xor(mx, 1));
            mx = fmaxf(mx, __shfl_xor(mx, 2));
            mx = fmaxf(mx, __shfl_xor(mx, 4));
            mx = fmaxf(mx, __shfl_xor(mx, 8));
            float newm = fmaxf(mrow[j], mx);
            float p0 = __expf(v0 - newm);
            float p1 = __expf(v1 - newm);
            float rs = p0 + p1;
            rs += __shfl_xor(rs, 1);
            rs += __shfl_xor(rs, 2);
            rs += __shfl_xor(rs, 4);
            rs += __shfl_xor(rs, 8);
            float alpha = __expf(mrow[j] - newm);
            lrow[j] = lrow[j] * alpha + rs;
            mrow[j] = newm;
            o[0][j] *= alpha; o[1][j] *= alpha; o[2][j] *= alpha; o[3][j] *= alpha;
            Plds[(8 * hi + j) * 32 + n]      = (_Float16)p0;
            Plds[(8 * hi + j) * 32 + 16 + n] = (_Float16)p1;
        }
        __syncthreads();
        Frag16 apv;
        const _Float16* pp = Plds + n * 32 + 16 * hi;
        apv.h[0] = *(const h8*)pp;
        apv.h[1] = *(const h8*)(pp + 8);
#pragma unroll
        for (int f = 0; f < 4; ++f) {
            Frag16 bv;
#if HAVE_TDM
            const _Float16* bp = &Vsh[buf][(16 * f + n) * 32 + 16 * hi];
#else
            const _Float16* bp = Vt + (size_t)(16 * f + n) * SEQ + k0 + 16 * hi;
#endif
            bv.h[0] = *(const h8*)bp;
            bv.h[1] = *(const h8*)(bp + 8);
            o[f] = __builtin_amdgcn_wmma_f32_16x16x32_f16(
                false, apv.v, false, bv.v, (short)0, o[f], false, false);
        }
        __syncthreads();
    }
#pragma unroll
    for (int j = 0; j < 8; ++j) {
        int s = q0 + 8 * hi + j;
        float inv = 1.0f / lrow[j];
        _Float16* orow = attn_out + ((size_t)(b * SEQ + s)) * DIMC + h * HDIM;
#pragma unroll
        for (int f = 0; f < 4; ++f)
            orow[16 * f + n] = (_Float16)(o[f][j] * inv);
    }
}

// ---------------- launcher ----------------
extern "C" void kernel_launch(void* const* d_in, const int* in_sizes, int n_in,
                              void* d_out, int out_size, void* d_ws, size_t ws_size,
                              hipStream_t stream) {
    (void)in_sizes; (void)n_in; (void)out_size; (void)ws_size;
    const float* x    = (const float*)d_in[0];
    const float* sinb = (const float*)d_in[1];
    const float* cosb = (const float*)d_in[2];
    const float* Wq   = (const float*)d_in[3];
    const float* Wk   = (const float*)d_in[4];
    const float* Wv   = (const float*)d_in[5];
    const float* Wo   = (const float*)d_in[6];
    const float* s_qk = (const float*)d_in[7];
    float* out = (float*)d_out;

    _Float16* x16  = (_Float16*)d_ws;
    _Float16* wq16 = x16  + (size_t)NROWS * DIMC;
    _Float16* wk16 = wq16 + (size_t)DIMC * DIMC;
    _Float16* wv16 = wk16 + (size_t)DIMC * DIMC;
    _Float16* wo16 = wv16 + (size_t)DIMC * DIMC;
    _Float16* q16  = wo16 + (size_t)DIMC * DIMC;               // [b,h,s,d]
    _Float16* k16  = q16  + (size_t)NROWS * DIMC;
    _Float16* vT   = k16  + (size_t)NROWS * DIMC;              // [b,h,d,s]
    _Float16* at16 = vT   + (size_t)NROWS * DIMC;              // [b,s,h*d]

    const int nx4 = NROWS * DIMC / 4, nw4 = DIMC * DIMC / 4;
    ngpt_f32_to_f16<<<(nx4 + 255) / 256, 256, 0, stream>>>((const float4*)x,  (h4*)x16,  nx4);
    ngpt_f32_to_f16<<<(nw4 + 255) / 256, 256, 0, stream>>>((const float4*)Wq, (h4*)wq16, nw4);
    ngpt_f32_to_f16<<<(nw4 + 255) / 256, 256, 0, stream>>>((const float4*)Wk, (h4*)wk16, nw4);
    ngpt_f32_to_f16<<<(nw4 + 255) / 256, 256, 0, stream>>>((const float4*)Wv, (h4*)wv16, nw4);
    ngpt_f32_to_f16<<<(nw4 + 255) / 256, 256, 0, stream>>>((const float4*)Wo, (h4*)wo16, nw4);

    dim3 gg(NROWS / 128, NHEAD), bb(256);
    ngpt_gemm_qk_rope<<<gg, bb, 0, stream>>>(x16, wq16, sinb, cosb, s_qk, q16);
    ngpt_gemm_qk_rope<<<gg, bb, 0, stream>>>(x16, wk16, sinb, cosb, s_qk, k16);
    ngpt_gemm_v    <<<gg, bb, 0, stream>>>(x16, wv16, vT);

    ngpt_flash_attn<<<dim3(SEQ / 16, BATCH * NHEAD), 32, 0, stream>>>(q16, k16, vT, at16);

    ngpt_gemm_out<<<dim3(NROWS / 128, DIMC / 64), bb, 0, stream>>>(at16, wo16, out);
}
